// OneLayerTransformer_14998025797885
// MI455X (gfx1250) — matrix operand order repaired and checked by reading, not compiled, not run on hardware
//
#include <hip/hip_runtime.h>

// Problem constants (match reference)
constexpr int N_OBS = 32;
constexpr int N_STATE = 64;
constexpr int HID = 128;
constexpr int BATCH = 2;
constexpr int SEQ = 8192;
constexpr int NSPLIT = 2;   // key-dimension split for occupancy

typedef __attribute__((ext_vector_type(16))) _Float16 v16h;
typedef __attribute__((ext_vector_type(8)))  _Float16 v8h;
typedef __attribute__((ext_vector_type(8)))  float    v8f;

static __device__ __forceinline__ v8f wmma_f16(v16h a, v16h b, v8f c) {
  // D = A(16x32 f16) * B(32x16 f16) + C(16x16 f32)
  return __builtin_amdgcn_wmma_f32_16x16x32_f16(false, a, false, b, (short)0, c, false, false);
}

static __device__ __forceinline__ v16h make_v16h(v8h lo, v8h hi) {
  v16h r;
#pragma unroll
  for (int i = 0; i < 8; ++i) { r[i] = lo[i]; r[i + 8] = hi[i]; }
  return r;
}

static __device__ __forceinline__ v8f splat8(float x) {
  v8f r;
#pragma unroll
  for (int i = 0; i < 8; ++i) r[i] = x;
  return r;
}

// ---------------------------------------------------------------------------
// Kernel 0: weight prep — transpose W1/W2 into f16 [out][in] rows so WMMA
// B-fragments are two contiguous 16B per-lane loads.
// ---------------------------------------------------------------------------
__global__ void kprep(const float* __restrict__ W1, const float* __restrict__ W2,
                      _Float16* __restrict__ W1t, _Float16* __restrict__ W2t) {
  int idx = blockIdx.x * blockDim.x + threadIdx.x;
  if (idx < HID * HID) {
    int o = idx & (HID - 1);
    int i = idx >> 7;
    W1t[(size_t)o * HID + i] = (_Float16)W1[(size_t)i * HID + o];
  }
  if (idx < HID * N_STATE) {
    int o = idx % N_STATE;
    int i = idx / N_STATE;
    W2t[(size_t)o * HID + i] = (_Float16)W2[(size_t)i * N_STATE + o];
  }
}

// ---------------------------------------------------------------------------
// Kernel 1: h = y @ W_obs + b_obs   (f32 in, f16 out). Tiny: VALU is fine.
// ---------------------------------------------------------------------------
__global__ void kproj1(const float* __restrict__ y, const float* __restrict__ Wobs,
                       const float* __restrict__ bobs, _Float16* __restrict__ hf) {
  int idx = blockIdx.x * blockDim.x + threadIdx.x;
  if (idx >= BATCH * SEQ * HID) return;
  int d = idx & (HID - 1);
  int row = idx >> 7;  // b*SEQ + t
  const float* yr = y + (size_t)row * N_OBS;
  float s = bobs[d];
#pragma unroll
  for (int k = 0; k < N_OBS; ++k) s += yr[k] * Wobs[(size_t)k * HID + d];
  hf[idx] = (_Float16)s;
}

// ---------------------------------------------------------------------------
// Kernel 2: Q/K/V projections. Q pre-scaled by HID^-0.5, K row-major,
// V stored transposed [b][d][t] for contiguous WMMA B-fragment loads.
// ---------------------------------------------------------------------------
__global__ void kproj2(const _Float16* __restrict__ hf,
                       const float* __restrict__ Wq, const float* __restrict__ bq,
                       const float* __restrict__ Wk, const float* __restrict__ bk,
                       const float* __restrict__ Wv, const float* __restrict__ bv,
                       _Float16* __restrict__ Qs, _Float16* __restrict__ Kf,
                       _Float16* __restrict__ Vt) {
  int idx = blockIdx.x * blockDim.x + threadIdx.x;
  if (idx >= BATCH * SEQ * HID) return;
  int d = idx & (HID - 1);
  int bt = idx >> 7;
  const _Float16* hr = hf + (size_t)bt * HID;
  float q = bq[d], k = bk[d], v = bv[d];
#pragma unroll 4
  for (int i = 0; i < HID; ++i) {
    float h = (float)hr[i];
    q += h * Wq[(size_t)i * HID + d];
    k += h * Wk[(size_t)i * HID + d];
    v += h * Wv[(size_t)i * HID + d];
  }
  Qs[idx] = (_Float16)(q * 0.08838834764831845f);  // 1/sqrt(128)
  Kf[idx] = (_Float16)k;
  int b = bt >> 13;          // / SEQ
  int t = bt & (SEQ - 1);
  Vt[((size_t)b * HID + d) * SEQ + t] = (_Float16)v;
}

// ---------------------------------------------------------------------------
// Kernel 3: causal flash attention, 2-way key-split partials.
// Scores computed TRANSPOSED (S^T = K * Q^T): the C-layout of S^T equals the
// A-fragment layout required by P*V, so exp(S) converts to the next WMMA's A
// operand purely in registers. No running max (scores are O(1) here), so
// split partials (sum exp(s)*V, sum exp(s)) combine linearly in kernel 4.
// Split s handles key blocks kb == 32*s (mod 64) -> balanced causal work.
// ---------------------------------------------------------------------------
__global__ void __launch_bounds__(128)
kattn_part(const _Float16* __restrict__ Qs, const _Float16* __restrict__ Kf,
           const _Float16* __restrict__ Vt, float* __restrict__ hacc,
           float* __restrict__ lsumP) {
  const int lane = threadIdx.x & 31;
  const int w    = threadIdx.x >> 5;
  const int hi16 = lane >> 4;
  const int lo16 = lane & 15;

  const int nq  = SEQ / 64;
  const int s   = blockIdx.x / (BATCH * nq);   // split id
  const int rem = blockIdx.x % (BATCH * nq);
  const int b   = rem / nq;
  const int q0  = (rem % nq) * 64 + w * 16;    // this wave's q-tile
  const long bt0 = (long)b * SEQ + q0;

  // ---- Q B-fragments (32x16: K=dims, N=queries), held for whole loop ----
  v16h bQ[4];
  {
    const _Float16* qrow = Qs + (size_t)(bt0 + lo16) * HID;
#pragma unroll
    for (int kc = 0; kc < 4; ++kc)
      bQ[kc] = make_v16h(*(const v8h*)(qrow + kc * 32 + hi16 * 8), *(const v8h*)(qrow + kc * 32 + hi16 * 8 + 16));
  }

  v8f acc[8];
#pragma unroll
  for (int dc = 0; dc < 8; ++dc) acc[dc] = splat8(0.f);
  float lsum = 0.f;

  for (int kb = s * 32; kb <= q0 + 15; kb += 64) {
    // prefetch this split's next key block (global_prefetch_b8)
    __builtin_prefetch((const void*)(Kf + ((size_t)b * SEQ + kb + 64 + lo16) * HID), 0, 3);

    // ---- S^T = K * Q^T : two 16x16 tiles (keys kb.., kb+16..) ----
    v8f st0 = splat8(0.f), st1 = splat8(0.f);
#pragma unroll
    for (int kc = 0; kc < 4; ++kc) {
      const _Float16* k0 = Kf + ((size_t)b * SEQ + kb + lo16) * HID + kc * 32 + hi16 * 8;
      v16h aK = make_v16h(*(const v8h*)k0, *(const v8h*)(k0 + 16));
      st0 = wmma_f16(aK, bQ[kc], st0);
    }
#pragma unroll
    for (int kc = 0; kc < 4; ++kc) {
      const _Float16* k1 = Kf + ((size_t)b * SEQ + kb + 16 + lo16) * HID + kc * 32 + hi16 * 8;
      v16h aK = make_v16h(*(const v8h*)k1, *(const v8h*)(k1 + 16));
      st1 = wmma_f16(aK, bQ[kc], st1);
    }

    // ---- causal mask: element = (key kb + [16] + r + 8*hi16, query q0 + lo16) ----
    if (kb + 15 > q0) {
#pragma unroll
      for (int r = 0; r < 8; ++r)
        if (kb + r + 8 * hi16 > q0 + lo16) st0[r] = -3.0e38f;
    }
    if (kb + 31 > q0) {
#pragma unroll
      for (int r = 0; r < 8; ++r)
        if (kb + 16 + r + 8 * hi16 > q0 + lo16) st1[r] = -3.0e38f;
    }

    // ---- P = exp(S): in-register conversion to the P*V A-fragment ----
    v16h aP;
    float ps = 0.f;
#pragma unroll
    for (int r = 0; r < 8; ++r) {
      float p0 = __expf(st0[r]);   // masked -> exp(-3e38) = 0
      float p1 = __expf(st1[r]);
      ps += p0 + p1;
      aP[r]     = (_Float16)p0;    // h=0..7  <-> keys hi16*8 + r
      aP[r + 8] = (_Float16)p1;    // h=8..15 <-> keys 16 + hi16*8 + r
    }
    lsum += ps;

    // ---- acc += P * V ----
#pragma unroll
    for (int dc = 0; dc < 8; ++dc) {
      const _Float16* vptr =
          Vt + ((size_t)b * HID + dc * 16 + lo16) * SEQ + kb + hi16 * 8;
      acc[dc] = wmma_f16(aP, make_v16h(*(const v8h*)vptr, *(const v8h*)(vptr + 16)), acc[dc]);
    }
  }

  // ---- write raw partials (plain [row][d] layout, coalesced over lo16) ----
  float ltot = lsum + __shfl_xor(lsum, 16, 32);
  float* ha = hacc + (size_t)s * BATCH * SEQ * HID;
#pragma unroll
  for (int dc = 0; dc < 8; ++dc)
#pragma unroll
    for (int r = 0; r < 8; ++r)
      ha[(size_t)(bt0 + r + 8 * hi16) * HID + dc * 16 + lo16] = acc[dc][r];
  if (hi16 == 0)
    lsumP[(size_t)s * BATCH * SEQ + bt0 + lo16] = ltot;
}

// ---------------------------------------------------------------------------
// Kernel 4: combine split partials, normalize, and fused MLP (all WMMA).
// ---------------------------------------------------------------------------
__global__ void __launch_bounds__(128)
kmlp(const float* __restrict__ hacc, const float* __restrict__ lsumP,
     const _Float16* __restrict__ W1t, const _Float16* __restrict__ W2t,
     const float* __restrict__ b1, const float* __restrict__ b2,
     float* __restrict__ out) {
  __shared__ __align__(16) _Float16 lds_h[4][16 * 136];
  __shared__ __align__(16) _Float16 lds_m[4][16 * 136];

  const int lane = threadIdx.x & 31;
  const int w    = threadIdx.x >> 5;
  const int hi16 = lane >> 4;
  const int lo16 = lane & 15;

  const int nq = SEQ / 64;
  const int b  = blockIdx.x / nq;
  const int q0 = (blockIdx.x % nq) * 64 + w * 16;
  const long bt0 = (long)b * SEQ + q0;

  const float* ha0 = hacc;
  const float* ha1 = hacc + (size_t)BATCH * SEQ * HID;
  const float* l0  = lsumP;
  const float* l1  = lsumP + (size_t)BATCH * SEQ;

  // ---- h_attn = (acc0+acc1) / (l0+l1) -> LDS f16 ----
#pragma unroll
  for (int r = 0; r < 8; ++r) {
    const int row = r + 8 * hi16;
    const float linv = 1.0f / (l0[bt0 + row] + l1[bt0 + row]);
#pragma unroll
    for (int dc = 0; dc < 8; ++dc) {
      const size_t o = (size_t)(bt0 + row) * HID + dc * 16 + lo16;
      lds_h[w][row * 136 + dc * 16 + lo16] = (_Float16)((ha0[o] + ha1[o]) * linv);
    }
  }
  __builtin_amdgcn_wave_barrier();

  v16h aH[4];
#pragma unroll
  for (int kc = 0; kc < 4; ++kc) {
    const _Float16* hrow = &lds_h[w][lo16 * 136 + kc * 32 + hi16 * 8];
    aH[kc] = make_v16h(*(const v8h*)(hrow), *(const v8h*)(hrow + 16));
  }

  // ---- mid = relu(h_attn @ W1 + b1) ----
#pragma unroll
  for (int oc = 0; oc < 8; ++oc) {
    v8f c = splat8(b1[oc * 16 + lo16]);
#pragma unroll
    for (int kc = 0; kc < 4; ++kc) {
      const _Float16* wp = W1t + (size_t)(oc * 16 + lo16) * HID + kc * 32 + hi16 * 8;
      c = wmma_f16(aH[kc], make_v16h(*(const v8h*)wp, *(const v8h*)(wp + 16)), c);
    }
#pragma unroll
    for (int r = 0; r < 8; ++r) {
      float m = fmaxf(c[r], 0.f);
      lds_m[w][(r + 8 * hi16) * 136 + oc * 16 + lo16] = (_Float16)m;
    }
  }
  __builtin_amdgcn_wave_barrier();

  v16h aM[4];
#pragma unroll
  for (int kc = 0; kc < 4; ++kc) {
    const _Float16* mr = &lds_m[w][lo16 * 136 + kc * 32 + hi16 * 8];
    aM[kc] = make_v16h(*(const v8h*)(mr), *(const v8h*)(mr + 16));
  }

  // ---- out = mid @ W2 + b2 ----
#pragma unroll
  for (int oc = 0; oc < 4; ++oc) {
    v8f c = splat8(b2[oc * 16 + lo16]);
#pragma unroll
    for (int kc = 0; kc < 4; ++kc) {
      const _Float16* wp = W2t + (size_t)(oc * 16 + lo16) * HID + kc * 32 + hi16 * 8;
      c = wmma_f16(aM[kc], make_v16h(*(const v8h*)wp, *(const v8h*)(wp + 16)), c);
    }
#pragma unroll
    for (int r = 0; r < 8; ++r) {
      out[(size_t)(bt0 + r + 8 * hi16) * N_STATE + oc * 16 + lo16] = c[r];
    }
  }
}

// ---------------------------------------------------------------------------
extern "C" void kernel_launch(void* const* d_in, const int* in_sizes, int n_in,
                              void* d_out, int out_size, void* d_ws, size_t ws_size,
                              hipStream_t stream) {
  (void)in_sizes; (void)n_in; (void)out_size; (void)ws_size;
  const float* y    = (const float*)d_in[0];
  const float* Wobs = (const float*)d_in[1];
  const float* bobs = (const float*)d_in[2];
  const float* Wq   = (const float*)d_in[3];
  const float* bq   = (const float*)d_in[4];
  const float* Wk   = (const float*)d_in[5];
  const float* bk   = (const float*)d_in[6];
  const float* Wv   = (const float*)d_in[7];
  const float* bv   = (const float*)d_in[8];
  const float* W1   = (const float*)d_in[9];
  const float* b1   = (const float*)d_in[10];
  const float* W2   = (const float*)d_in[11];
  const float* b2   = (const float*)d_in[12];
  float* out = (float*)d_out;

  const size_t n = (size_t)BATCH * SEQ * HID;
  _Float16* Qs  = (_Float16*)d_ws;
  _Float16* Kf  = Qs + n;
  _Float16* Vt  = Kf + n;
  _Float16* hf  = Vt + n;
  _Float16* W1t = hf + n;
  _Float16* W2t = W1t + (size_t)HID * HID;
  float* hacc   = (float*)(W2t + (size_t)HID * N_STATE);     // NSPLIT * n floats
  float* lsumP  = hacc + (size_t)NSPLIT * n;                 // NSPLIT * BATCH*SEQ

  kprep<<<(HID * HID + 255) / 256, 256, 0, stream>>>(W1, W2, W1t, W2t);
  kproj1<<<(int)((n + 255) / 256), 256, 0, stream>>>(y, Wobs, bobs, hf);
  kproj2<<<(int)((n + 255) / 256), 256, 0, stream>>>(hf, Wq, bq, Wk, bk, Wv, bv, Qs, Kf, Vt);
  kattn_part<<<NSPLIT * BATCH * (SEQ / 64), 128, 0, stream>>>(Qs, Kf, Vt, hacc, lsumP);
  kmlp<<<BATCH * (SEQ / 64), 128, 0, stream>>>(hacc, lsumP, W1t, W2t, b1, b2, out);
}
